// StructuredTransformerDecoderBlock_11467562680630
// MI455X (gfx1250) — matrix operand order repaired and checked by reading, not compiled, not run on hardware
//
#include <hip/hip_runtime.h>

// ---------------------------------------------------------------------------
// StructuredTransformerDecoderBlock for MI455X (gfx1250, wave32, WMMA f16)
//
// N=50000 nodes, K=16 neighbours, feat=128, msg=192, heads=8, att=32.
// - logits = msg . (Wk_h @ q)  restructure (kills the 39-GMAC k-projection)
// - f16 WMMA (16x16x32) with f32 accumulation for all dense GEMMs
// - weights pre-converted to f16 *transposed* (N-major) so each B fragment
//   is two contiguous global_load_b128 per lane (no strided half gathers)
// ---------------------------------------------------------------------------

typedef __attribute__((ext_vector_type(16))) _Float16 v16h;
typedef __attribute__((ext_vector_type(8)))  _Float16 v8h;
typedef __attribute__((ext_vector_type(8)))  float    v8f;
typedef __attribute__((ext_vector_type(2)))  _Float16 h2v;

#define NNODES   50000
#define KNBR     16
#define FSIZE    128
#define DDIM     32
#define SDIM     32
#define IN_DIM   192
#define HEADS    8
#define ATT      32
#define SCALE_F  0.17677669529663687f   // 1/sqrt(32)

// ---- f16 weight workspace offsets (element counts); all stored N-major ----
#define OFF_WQT  0        // WqT  [256][128]   WqT[n][k] = Wq[k][n]
#define OFF_WK   32768    // Wk   [192][256]   row-major copy (acts as BT for P)
#define OFF_WVT  81920    // WvT  [1024][192]  WvT[c][k] = Wv[k][c]
#define OFF_WOT  278528   // WoT  [128][1024]  WoT[n][k] = Wo[k][n]
#define OFF_W1T  409600   // W1T  [128][128]
#define OFF_W2T  425984   // W2T  [128][128]
#define OFF_W3T  442368   // W3T  [128][128]
#define TOTAL_W  458752

// ---- LDS layout (bytes); strides in _Float16/float elements ----
#define MSG_STR  200      // 192 + pad (multiple of 8 for aligned v8h loads)
#define PH_STR   1552     // 8*192 + pad
#define VH_STR   1040     // 1024 + pad
#define XH_STR   136
#define XF_STR   132

#define OFF_MSG  0        // 256 x MSG_STR f16 = 102400   (dead after m-stage)
#define OFF_VH   0        //  16 x VH_STR  f16 =  33280   (overlays MSG)
#define OFF_PH   102400   //  16 x PH_STR  f16 =  49664   (dead after logits)
#define OFF_MH   102400   // 128 x MSG_STR f16 =  51200   (overlays PH)
#define OFF_XH   153600   //  16 x XH_STR  f16 =   4352
#define OFF_XF   157952   //  16 x XF_STR  f32 =   8448
#define OFF_QH   166400   //  16 x 264     f16 =   8448
#define OFF_QBK  174848   // 128 f32            =    512
#define OFF_LG   175360   // 256 x 8 f32        =   8192
#define OFF_O1   183552   //  16 x XF_STR  f32 =   8448
#define OFF_O1H  192000   //  16 x XH_STR  f16 =   4352
#define OFF_H1   196352   //  16 x XH_STR  f16 =   4352
#define OFF_H2   200704   //  16 x XH_STR  f16 =   4352
#define SMEM_BYTES 205056

// ---------------- WMMA fragment helpers (CDNA5 wave32 layouts) -------------
// A (16x32 f16) from row-major LDS [row][k], stride in halves (mult of 8):
//   lane = hf*16 + r ; elements j<8 -> A[r][hf*8+j] ; j>=8 -> A[r][16+hf*8+j-8]
__device__ __forceinline__ v16h load_A_f16(const _Float16* base, int stride, int lane) {
  const int hf = lane >> 4, r = lane & 15;
  const _Float16* p = base + r * stride + hf * 8;
  v8h lo = *(const v8h*)p;          // K = hf*8 + 0..7
  v8h hi = *(const v8h*)(p + 16);   // K = 16 + hf*8 + 0..7
  v16h a;
#pragma unroll
  for (int j = 0; j < 8; ++j) { a[j] = lo[j]; a[8 + j] = hi[j]; }
  return a;
}

// B (32x16 f16) from transposed (N-major) weights BT[n][k], Kstride halves:
//   lane = hf*16 + c ; element j<8 -> B[hf*8+j][c] ; j>=8 -> B[16+hf*8+j-8][c]
//   -> two 8-half runs per lane = two 16-byte loads.
__device__ __forceinline__ v16h load_BT_f16(const _Float16* basN, int Kstride, int lane) {
  const int hf = lane >> 4, c = lane & 15;
  const _Float16* p = basN + c * Kstride + hf * 8;
  v8h lo = *(const v8h*)p;
  v8h hi = *(const v8h*)(p + 16);
  v16h b;
#pragma unroll
  for (int j = 0; j < 8; ++j) { b[j] = lo[j]; b[8 + j] = hi[j]; }
  return b;
}

__device__ __forceinline__ v8f wmma16(const v16h& a, const v16h& b, const v8f& c) {
  return __builtin_amdgcn_wmma_f32_16x16x32_f16(false, a, false, b, (short)0, c, false, false);
}

// ------------------- weight prep (f32 -> f16, transpose) -------------------
__global__ __launch_bounds__(256) void prep_weights(
    const float* __restrict__ Wq, const float* __restrict__ Wk,
    const float* __restrict__ Wv, const float* __restrict__ Wo,
    const float* __restrict__ W1, const float* __restrict__ W2,
    const float* __restrict__ W3, _Float16* __restrict__ w16) {
  int idx = blockIdx.x * 256 + threadIdx.x;
  if (idx >= TOTAL_W) return;
  float v;
  if (idx < OFF_WK)        { int n = idx >> 7, k = idx & 127; v = Wq[k * 256 + n]; }
  else if (idx < OFF_WVT)  { v = Wk[idx - OFF_WK]; }
  else if (idx < OFF_WOT)  { int t = idx - OFF_WVT; int c = t / IN_DIM, k = t - c * IN_DIM;
                             v = Wv[k * 1024 + c]; }
  else if (idx < OFF_W1T)  { int t = idx - OFF_WOT; int n = t >> 10, k = t & 1023;
                             v = Wo[k * 128 + n]; }
  else if (idx < OFF_W2T)  { int t = idx - OFF_W1T; int n = t >> 7, k = t & 127;
                             v = W1[k * 128 + n]; }
  else if (idx < OFF_W3T)  { int t = idx - OFF_W2T; int n = t >> 7, k = t & 127;
                             v = W2[k * 128 + n]; }
  else                     { int t = idx - OFF_W3T; int n = t >> 7, k = t & 127;
                             v = W3[k * 128 + n]; }
  w16[idx] = (_Float16)v;
}

// ------------------------------- main kernel -------------------------------
__global__ __launch_bounds__(256) void fused_block(
    const float* __restrict__ feat, const int* __restrict__ nbr,
    const float* __restrict__ distf, const float* __restrict__ seqf,
    const float* __restrict__ bq, const float* __restrict__ bk,
    const float* __restrict__ bv, const float* __restrict__ bo,
    const float* __restrict__ b1, const float* __restrict__ b2,
    const float* __restrict__ b3, const _Float16* __restrict__ w16,
    float* __restrict__ out) {
  extern __shared__ char smem[];
  _Float16* MSG = (_Float16*)(smem + OFF_MSG);
  _Float16* VH  = (_Float16*)(smem + OFF_VH);
  _Float16* PH  = (_Float16*)(smem + OFF_PH);
  _Float16* MH  = (_Float16*)(smem + OFF_MH);
  _Float16* XH  = (_Float16*)(smem + OFF_XH);
  float*    XF  = (float*)   (smem + OFF_XF);
  _Float16* QH  = (_Float16*)(smem + OFF_QH);
  float*    QBK = (float*)   (smem + OFF_QBK);
  float*    LG  = (float*)   (smem + OFF_LG);
  float*    O1  = (float*)   (smem + OFF_O1);
  _Float16* O1H = (_Float16*)(smem + OFF_O1H);
  _Float16* H1  = (_Float16*)(smem + OFF_H1);
  _Float16* H2  = (_Float16*)(smem + OFF_H2);

  const int tid  = threadIdx.x;
  const int lane = tid & 31;
  const int wv   = tid >> 5;                 // wave id, 0..7
  const int gbase = blockIdx.x * 16;         // 3125 * 16 == 50000 exactly
  const int hf = lane >> 4, cc = lane & 15;  // D-fragment row-half / column

  // ---- stage 0: load X tile (f32 residual copy + f16 GEMM copy) ----
  for (int r = wv; r < 16; r += 8) {
    const float* src = feat + (size_t)(gbase + r) * FSIZE;
    for (int i = lane; i < FSIZE; i += 32) {
      float f = src[i];
      XF[r * XF_STR + i] = f;
      XH[r * XH_STR + i] = (_Float16)f;
    }
  }
  // ---- stage 0b: gather neighbour messages -> MSG f16 (256 rows x 192) ----
  for (int r = wv; r < 256; r += 8) {
    int n = r >> 4, k = r & 15;
    int nb = nbr[(size_t)(gbase + n) * KNBR + k];
    _Float16* m = MSG + r * MSG_STR;
    const float* fs = feat + (size_t)nb * FSIZE;
    for (int i = lane; i < FSIZE; i += 32) m[i] = (_Float16)fs[i];
    const float* ds = distf + ((size_t)(gbase + n) * KNBR + k) * DDIM;
    for (int i = lane; i < DDIM; i += 32) m[FSIZE + i] = (_Float16)ds[i];
    const float* ss = seqf + ((size_t)(gbase + n) * KNBR + k) * SDIM;
    for (int i = lane; i < SDIM; i += 32) m[FSIZE + DDIM + i] = (_Float16)ss[i];
  }
  __syncthreads();

  // ---- stage 1: Q = X @ Wq + bq   (M=16, K=128, N=256) ----
  for (int nt = wv; nt < 16; nt += 8) {
    v8f acc = {};
#pragma unroll
    for (int ks = 0; ks < 4; ++ks) {
      v16h a = load_A_f16(XH + ks * 32, XH_STR, lane);
      v16h b = load_BT_f16(w16 + OFF_WQT + (nt * 16) * 128 + ks * 32, 128, lane);
      acc = wmma16(a, b, acc);
    }
    int col = nt * 16 + cc; float bias = bq[col];
#pragma unroll
    for (int r = 0; r < 8; ++r) QH[(r + hf * 8) * 264 + col] = (_Float16)(acc[r] + bias);
  }
  __syncthreads();

  // ---- stage 2a: qbk[n,h] = q[n,h,:] . bk[h,:]  (bias term of logits) ----
  if (tid < 128) {
    int n = tid >> 3, h = tid & 7; float s = 0.f;
    for (int a = 0; a < ATT; ++a) s += (float)QH[n * 264 + h * 32 + a] * bk[h * 32 + a];
    QBK[tid] = s;
  }
  // ---- stage 2b: P_h = Q_h @ Wk_h^T  (per head: M=16, K=32, N=192) ----
  // BT for this GEMM is row-major Wk itself: BT[i][a] = Wk[i][h*32+a].
  for (int task = wv; task < 96; task += 8) {
    int h = task / 12, nt = task % 12;
    v16h a = load_A_f16(QH + h * 32, 264, lane);
    v16h b = load_BT_f16(w16 + OFF_WK + (nt * 16) * 256 + h * 32, 256, lane);
    v8f acc = {};
    acc = wmma16(a, b, acc);
#pragma unroll
    for (int r = 0; r < 8; ++r)
      PH[(r + hf * 8) * PH_STR + h * IN_DIM + nt * 16 + cc] = (_Float16)acc[r];
  }
  __syncthreads();

  // ---- stage 3: logits[n,k,h] = SCALE*(msg . p + qbk) ----
  for (int t = tid; t < 2048; t += 256) {
    int h = t & 7, kk = (t >> 3) & 15, n = t >> 7;
    const h2v* m2 = (const h2v*)(MSG + (n * 16 + kk) * MSG_STR);
    const h2v* p2 = (const h2v*)(PH + n * PH_STR + h * IN_DIM);
    float s = 0.f;
    for (int i = 0; i < IN_DIM / 2; ++i) {
      h2v a = m2[i], b = p2[i];
      s += (float)a[0] * (float)b[0] + (float)a[1] * (float)b[1];
    }
    LG[(n * 16 + kk) * 8 + h] = (s + QBK[n * 8 + h]) * SCALE_F;
  }
  __syncthreads();

  // ---- stage 4: softmax over K=16, one thread per (n,h) ----
  if (tid < 128) {
    int n = tid >> 3, h = tid & 7;
    float l[16], mx = -1e30f;
    for (int k = 0; k < 16; ++k) { l[k] = LG[(n * 16 + k) * 8 + h]; mx = fmaxf(mx, l[k]); }
    float s = 0.f;
    for (int k = 0; k < 16; ++k) { l[k] = __expf(l[k] - mx); s += l[k]; }
    float inv = 1.f / s;
    for (int k = 0; k < 16; ++k) LG[(n * 16 + k) * 8 + h] = l[k] * inv;
  }
  __syncthreads();

  // ---- stage 5: m[n,h,:] = sum_k w[n,k,h] * msg[n,k,:]  (VALU) ----
  {
    int row = tid >> 1, t2 = tid & 1;        // 128 rows (n*8+h), 2 threads/row
    int n = row >> 3, h = row & 7;
    float wl[16];
#pragma unroll
    for (int k = 0; k < 16; ++k) wl[k] = LG[(n * 16 + k) * 8 + h];
    for (int c = t2 * 96; c < t2 * 96 + 96; ++c) {
      float s = 0.f;
#pragma unroll
      for (int k = 0; k < 16; ++k) s += wl[k] * (float)MSG[(n * 16 + k) * MSG_STR + c];
      MH[row * MSG_STR + c] = (_Float16)s;
    }
  }
  __syncthreads();

  // ---- stage 6: V_h = M_h @ Wv_h + bv  (per head: M=16, K=192, N=128) ----
  for (int task = wv; task < 64; task += 8) {
    int h = task >> 3, nt = task & 7;
    v8f acc = {};
#pragma unroll
    for (int ks = 0; ks < 6; ++ks) {
      v16h a = load_A_f16(MH + h * MSG_STR + ks * 32, 8 * MSG_STR, lane);
      v16h b = load_BT_f16(w16 + OFF_WVT + (h * 128 + nt * 16) * IN_DIM + ks * 32,
                           IN_DIM, lane);
      acc = wmma16(a, b, acc);
    }
    int col = h * 128 + nt * 16 + cc; float bias = bv[col];
#pragma unroll
    for (int r = 0; r < 8; ++r) VH[(r + hf * 8) * VH_STR + col] = (_Float16)(acc[r] + bias);
  }
  __syncthreads();

  // ---- stage 7: att = V @ Wo + bo ; out1 = x + att  (M=16, K=1024, N=128) ----
  {
    int nt = wv;
    v8f acc = {};
    for (int ks = 0; ks < 32; ++ks) {
      v16h a = load_A_f16(VH + ks * 32, VH_STR, lane);
      v16h b = load_BT_f16(w16 + OFF_WOT + (nt * 16) * 1024 + ks * 32, 1024, lane);
      acc = wmma16(a, b, acc);
    }
    int col = nt * 16 + cc; float bias = bo[col];
#pragma unroll
    for (int r = 0; r < 8; ++r) {
      int row = r + hf * 8;
      float v = acc[r] + bias + XF[row * XF_STR + col];
      O1[row * XF_STR + col]  = v;
      O1H[row * XH_STR + col] = (_Float16)v;
    }
  }
  __syncthreads();

  // ---- stage 8: MLP layer 1 (relu) ----
  {
    int nt = wv; v8f acc = {};
#pragma unroll
    for (int ks = 0; ks < 4; ++ks) {
      v16h a = load_A_f16(O1H + ks * 32, XH_STR, lane);
      v16h b = load_BT_f16(w16 + OFF_W1T + (nt * 16) * 128 + ks * 32, 128, lane);
      acc = wmma16(a, b, acc);
    }
    int col = nt * 16 + cc; float bias = b1[col];
#pragma unroll
    for (int r = 0; r < 8; ++r)
      H1[(r + hf * 8) * XH_STR + col] = (_Float16)fmaxf(acc[r] + bias, 0.f);
  }
  __syncthreads();
  // ---- stage 9: MLP layer 2 (relu) ----
  {
    int nt = wv; v8f acc = {};
#pragma unroll
    for (int ks = 0; ks < 4; ++ks) {
      v16h a = load_A_f16(H1 + ks * 32, XH_STR, lane);
      v16h b = load_BT_f16(w16 + OFF_W2T + (nt * 16) * 128 + ks * 32, 128, lane);
      acc = wmma16(a, b, acc);
    }
    int col = nt * 16 + cc; float bias = b2[col];
#pragma unroll
    for (int r = 0; r < 8; ++r)
      H2[(r + hf * 8) * XH_STR + col] = (_Float16)fmaxf(acc[r] + bias, 0.f);
  }
  __syncthreads();
  // ---- stage 10: MLP layer 3 + residual -> global ----
  {
    int nt = wv; v8f acc = {};
#pragma unroll
    for (int ks = 0; ks < 4; ++ks) {
      v16h a = load_A_f16(H2 + ks * 32, XH_STR, lane);
      v16h b = load_BT_f16(w16 + OFF_W3T + (nt * 16) * 128 + ks * 32, 128, lane);
      acc = wmma16(a, b, acc);
    }
    int col = nt * 16 + cc; float bias = b3[col];
#pragma unroll
    for (int r = 0; r < 8; ++r) {
      int row = r + hf * 8;
      out[(size_t)(gbase + row) * FSIZE + col] =
          acc[r] + bias + O1[row * XF_STR + col];
    }
  }
}

// ------------------------------- launcher ----------------------------------
extern "C" void kernel_launch(void* const* d_in, const int* in_sizes, int n_in,
                              void* d_out, int out_size, void* d_ws, size_t ws_size,
                              hipStream_t stream) {
  (void)in_sizes; (void)n_in; (void)out_size; (void)ws_size;
  const float* feat  = (const float*)d_in[0];
  const int*   nbrs  = (const int*)  d_in[1];
  const float* distf = (const float*)d_in[2];
  const float* seqf  = (const float*)d_in[3];
  const float* Wq = (const float*)d_in[4];  const float* bq = (const float*)d_in[5];
  const float* Wk = (const float*)d_in[6];  const float* bk = (const float*)d_in[7];
  const float* Wv = (const float*)d_in[8];  const float* bv = (const float*)d_in[9];
  const float* Wo = (const float*)d_in[10]; const float* bo = (const float*)d_in[11];
  const float* W1 = (const float*)d_in[12]; const float* b1 = (const float*)d_in[13];
  const float* W2 = (const float*)d_in[14]; const float* b2 = (const float*)d_in[15];
  const float* W3 = (const float*)d_in[16]; const float* b3 = (const float*)d_in[17];
  _Float16* w16 = (_Float16*)d_ws;          // 458752 halves = 896 KB scratch

  prep_weights<<<(TOTAL_W + 255) / 256, 256, 0, stream>>>(Wq, Wk, Wv, Wo, W1, W2, W3, w16);
  fused_block<<<NNODES / 16, 256, SMEM_BYTES, stream>>>(
      feat, nbrs, distf, seqf, bq, bk, bv, bo, b1, b2, b3, w16, (float*)d_out);
}
